// ProtoNet_8864812499469
// MI455X (gfx1250) — matrix operand order repaired and checked by reading, not compiled, not run on hardware
//
#include <hip/hip_runtime.h>
#include <hip/hip_bf16.h>
#include <math.h>

// ---------------- types / helpers -------------------------------------------
typedef __bf16 bf16_t;
typedef __attribute__((ext_vector_type(16))) __bf16 v16bf;
typedef __attribute__((ext_vector_type(8)))  float  v8f;

#define DFEAT  320
#define TOTR   65536            // 1024 same rows + 64*1008 diff rows
#define TOTRP  (TOTR + 32)      // padded V^T stride (zeroed pad)
#define KT_N   10               // 320/32 k-steps
#define NT_N   20               // 320/16 n-tiles

// native f32 -> bf16 convert: lets the backend emit v_cvt_*bf16_f32 instead of
// a 3-op integer RNE sequence per element
__device__ __forceinline__ bf16_t f2bf(float f) { return (bf16_t)f; }

__device__ __forceinline__ bf16_t bf_zero() {
  union { unsigned short s; bf16_t b; } r; r.s = 0; return r.b;
}

// A-fragment k index: lanes 0-15 K=0-15, 16-31 K=16-31 (matches kpatB)
__device__ __forceinline__ int kpatA(int e, int hf) {
  return (hf << 4) + e;
}
// B-fragment k index (contiguous order, same as kpatA): lanes 0-15 K=0-15, 16-31 K=16-31
__device__ __forceinline__ int kpatB(int e, int hf) { return (hf << 4) + e; }

__device__ __forceinline__ v8f wmma_bf16(v16bf a, v16bf b, v8f c) {
  return __builtin_amdgcn_wmma_f32_16x16x32_bf16(false, a, false, b,
                                                 (short)0, c, false, false);
}

// ---------------- kernel 0: pack 6 DxD weights into bf16 B-fragments --------
// layout: pack[mat][kt][nt][lane][e] , 16 contiguous bf16 (32B) per lane
__global__ void pack_weights_kernel(const float* w0, const float* w1,
                                    const float* w2, const float* w3,
                                    const float* w4, const float* w5,
                                    bf16_t* pack) {
  int tid = blockIdx.x * blockDim.x + threadIdx.x;
  if (tid >= 6 * KT_N * NT_N * 32) return;
  int mat  = tid / (KT_N * NT_N * 32);
  int rem  = tid % (KT_N * NT_N * 32);
  int kt   = rem / (NT_N * 32);
  int rem2 = rem % (NT_N * 32);
  int nt   = rem2 / 32;
  int lane = rem2 % 32;
  const float* w = (mat == 0) ? w0 : (mat == 1) ? w1 : (mat == 2) ? w2
                 : (mat == 3) ? w3 : (mat == 4) ? w4 : w5;
  bf16_t* dst = pack + (long)mat * DFEAT * DFEAT
                     + ((long)(kt * NT_N + nt) * 32 + lane) * 16;
  int hf = lane >> 4, n = nt * 16 + (lane & 15);
#pragma unroll
  for (int e = 0; e < 16; ++e) {
    int k = kt * 32 + kpatB(e, hf);
    dst[e] = f2bf(w[(long)k * DFEAT + n]);
  }
}

// zero the 32-row pad of V^T so attention tail tiles multiply garbage-free
__global__ void init_vt_pad_kernel(bf16_t* Vt) {
  int tid = blockIdx.x * blockDim.x + threadIdx.x;
  if (tid >= DFEAT * 32) return;
  int d = tid >> 5, r = tid & 31;
  Vt[(long)d * TOTRP + TOTR + r] = bf_zero();
}

// ---------------- kernel 1: P = Diff @ W{same,diff} + b - mean --------------
__global__ __launch_bounds__(32)
void gemm_p_kernel(const float* xmean, const float* x, const bf16_t* pack,
                   const float* b_same, const float* b_diff, float* P) {
  __shared__ float sx[16][DFEAT];
  __shared__ float smean[DFEAT];
  __shared__ float sbias[DFEAT];
  int lane = threadIdx.x;
  int mt = blockIdx.x, ng = blockIdx.y;
  long M0 = (long)mt * 16;

  int c; const bf16_t* wp; const float* bias; bool same; long jbase;
  if (M0 < 1024) { same = true;  c = mt; wp = pack; bias = b_same; jbase = 0; }
  else {
    long r0 = M0 - 1024; c = (int)(r0 / 1008); same = false;
    wp = pack + DFEAT * DFEAT; bias = b_diff; jbase = r0 - (long)c * 1008;
  }
  for (int i = lane; i < DFEAT; i += 32) {
    smean[i] = xmean[c * DFEAT + i];
    sbias[i] = bias[i];
  }
  // stage the 16 difference rows in LDS
  for (int i = lane; i < 16 * DFEAT; i += 32) {
    int row = i / DFEAT, k = i % DFEAT;
    float val;
    if (same) {
      int samp = c * 16 + row;
      val = x[(long)samp * DFEAT + k] - xmean[c * DFEAT + k];
    } else {
      long j = jbase + row;
      int o = (int)(j >> 4);
      int other = o + (o >= c);
      int samp = other * 16 + (int)(j & 15);
      val = xmean[c * DFEAT + k] - x[(long)samp * DFEAT + k];
    }
    sx[row][k] = val;
  }
  __syncthreads();

  int m = lane & 15, hf = lane >> 4;
  v8f acc[4] = {{}, {}, {}, {}};
  for (int kt = 0; kt < KT_N; ++kt) {
    v16bf a;
#pragma unroll
    for (int e = 0; e < 16; ++e) a[e] = f2bf(sx[m][kt * 32 + kpatA(e, hf)]);
#pragma unroll
    for (int t = 0; t < 4; ++t) {
      int nt = ng * 4 + t;
      const v16bf* bp =
          (const v16bf*)(wp + ((long)(kt * NT_N + nt) * 32 + lane) * 16);
      acc[t] = wmma_bf16(a, *bp, acc[t]);
    }
  }
#pragma unroll
  for (int t = 0; t < 4; ++t) {
    int nt = ng * 4 + t, n = nt * 16 + m;
#pragma unroll
    for (int r = 0; r < 8; ++r) {
      long row = M0 + r + 8 * hf;
      P[row * DFEAT + n] = acc[t][r] + sbias[n] - smean[n];
    }
  }
}

// ---------------- kernel 2: Q/K/V projections (z = 0/1/2) -------------------
__global__ __launch_bounds__(32)
void gemm_qkv_kernel(const float* P, const bf16_t* pack, const float* bq,
                     const float* bk, const float* bv, bf16_t* Q, bf16_t* K,
                     bf16_t* Vt) {
  int lane = threadIdx.x;
  int mt = blockIdx.x, ng = blockIdx.y, z = blockIdx.z;
  long M0 = (long)mt * 16;
  const bf16_t* wp = pack + (long)(2 + z) * DFEAT * DFEAT;
  const float* bias = (z == 0) ? bq : (z == 1) ? bk : bv;
  int m = lane & 15, hf = lane >> 4;
  const float* arow = P + (M0 + m) * DFEAT;
  v8f acc[4] = {{}, {}, {}, {}};
  for (int kt = 0; kt < KT_N; ++kt) {
    v16bf a;
#pragma unroll
    for (int e = 0; e < 16; ++e) a[e] = f2bf(arow[kt * 32 + kpatA(e, hf)]);
#pragma unroll
    for (int t = 0; t < 4; ++t) {
      int nt = ng * 4 + t;
      const v16bf* bp =
          (const v16bf*)(wp + ((long)(kt * NT_N + nt) * 32 + lane) * 16);
      acc[t] = wmma_bf16(a, *bp, acc[t]);
    }
  }
#pragma unroll
  for (int t = 0; t < 4; ++t) {
    int nt = ng * 4 + t, n = nt * 16 + m;
#pragma unroll
    for (int r = 0; r < 8; ++r) {
      long row = M0 + r + 8 * hf;
      float v = acc[t][r] + bias[n];
      if (z == 0)      Q[row * DFEAT + n] = f2bf(v);
      else if (z == 1) K[row * DFEAT + n] = f2bf(v);
      else             Vt[(long)n * TOTRP + row] = f2bf(v);   // transposed V
    }
  }
}

// ---------------- kernel 3: per-class attention (seq 16 or 1008) ------------
__global__ __launch_bounds__(32)
void attn_kernel(const bf16_t* Q, const bf16_t* K, const bf16_t* Vt,
                 bf16_t* O) {
  __shared__ float S[16][1024];                  // 64 KB score block
  int lane = threadIdx.x;
  int m = lane & 15, hf = lane >> 4;
  int c = blockIdx.x >> 6, tt = blockIdx.x & 63;
  long Q0, K0; int Lk;
  if (tt == 0) { Q0 = (long)c * 16; K0 = Q0; Lk = 16; }
  else {
    K0 = 1024 + (long)c * 1008;
    Q0 = K0 + (long)(tt - 1) * 16;
    Lk = 1008;
  }
  const float scale = 0.05590169943749474f;      // 1/sqrt(320)

  // phase 1: S = Q K^T * scale
  int nkt = Lk >> 4;
  const bf16_t* qrow = Q + (Q0 + m) * DFEAT;
  for (int kt2 = 0; kt2 < nkt; ++kt2) {
    v8f s = {};
    const bf16_t* krow = K + (K0 + kt2 * 16 + m) * DFEAT;
    for (int kc = 0; kc < KT_N; ++kc) {
      v16bf a, b;
#pragma unroll
      for (int e = 0; e < 16; ++e) a[e] = qrow[kc * 32 + kpatA(e, hf)];
      b = *(const v16bf*)(krow + kc * 32 + hf * 16);   // contiguous kpatB
      s = wmma_bf16(a, b, s);
    }
#pragma unroll
    for (int r = 0; r < 8; ++r)
      S[r + 8 * hf][kt2 * 16 + m] = s[r] * scale;
  }
  __syncthreads();

  // phase 2: row softmax, all 32 lanes (2 lanes per row), + zero pad to 1024
  {
    int row = lane >> 1, h2 = lane & 1;
    float mx = -3.4e38f;
    for (int j = h2; j < Lk; j += 2) mx = fmaxf(mx, S[row][j]);
    mx = fmaxf(mx, __shfl_xor(mx, 1));
    float sum = 0.0f;
    for (int j = h2; j < Lk; j += 2) {
      float e = __expf(S[row][j] - mx); S[row][j] = e; sum += e;
    }
    sum += __shfl_xor(sum, 1);
    float inv = 1.0f / sum;
    for (int j = h2; j < Lk; j += 2) S[row][j] *= inv;
    for (int j = Lk + h2; j < 1024; j += 2) S[row][j] = 0.0f;
  }
  __syncthreads();

  // phase 3: O = softmax(S) @ V
  int nkc = (Lk == 16) ? 1 : 32;
  for (int nt = 0; nt < NT_N; ++nt) {
    v8f acc = {};
    int dcol = nt * 16 + m;
    const bf16_t* vrow = Vt + (long)dcol * TOTRP + K0;
    for (int kc = 0; kc < nkc; ++kc) {
      v16bf a, b;
#pragma unroll
      for (int e = 0; e < 16; ++e) a[e] = f2bf(S[m][kc * 32 + kpatA(e, hf)]);
      b = *(const v16bf*)(vrow + kc * 32 + hf * 16);   // pad rows are zero
      acc = wmma_bf16(a, b, acc);
    }
#pragma unroll
    for (int r = 0; r < 8; ++r)
      O[(Q0 + r + 8 * hf) * DFEAT + nt * 16 + m] = f2bf(acc[r]);
  }
}

// ---------------- kernel 4: H = O @ Wfc + bfc + P (residual) ----------------
__global__ __launch_bounds__(32)
void gemm_fc_kernel(const bf16_t* O, const bf16_t* pack, const float* bfc,
                    const float* P, float* H) {
  int lane = threadIdx.x;
  int mt = blockIdx.x, ng = blockIdx.y;
  long M0 = (long)mt * 16;
  const bf16_t* wp = pack + (long)5 * DFEAT * DFEAT;
  int m = lane & 15, hf = lane >> 4;
  const bf16_t* arow = O + (M0 + m) * DFEAT;
  v8f acc[4] = {{}, {}, {}, {}};
  for (int kt = 0; kt < KT_N; ++kt) {
    v16bf a;
#pragma unroll
    for (int e = 0; e < 16; ++e) a[e] = arow[kt * 32 + kpatA(e, hf)];
#pragma unroll
    for (int t = 0; t < 4; ++t) {
      int nt = ng * 4 + t;
      const v16bf* bp =
          (const v16bf*)(wp + ((long)(kt * NT_N + nt) * 32 + lane) * 16);
      acc[t] = wmma_bf16(a, *bp, acc[t]);
    }
  }
#pragma unroll
  for (int t = 0; t < 4; ++t) {
    int nt = ng * 4 + t, n = nt * 16 + m;
#pragma unroll
    for (int r = 0; r < 8; ++r) {
      long row = M0 + r + 8 * hf;
      H[row * DFEAT + n] = acc[t][r] + bfc[n] + P[row * DFEAT + n];
    }
  }
}

// ---------------- kernel 5: LayerNorm + logit per row (wave per row) --------
__global__ void ln_logits_kernel(const float* H, const float* g,
                                 const float* bta, const float* Wout,
                                 const float* bout, float* logits) {
  int w = threadIdx.x >> 5, lane = threadIdx.x & 31;
  long row = (long)blockIdx.x * 8 + w;
  const float* h = H + row * DFEAT;
  float s = 0.0f;
  for (int i = lane; i < DFEAT; i += 32) s += h[i];
  for (int m = 16; m; m >>= 1) s += __shfl_xor(s, m);
  float mu = s * (1.0f / DFEAT);
  float v = 0.0f;
  for (int i = lane; i < DFEAT; i += 32) { float d = h[i] - mu; v += d * d; }
  for (int m = 16; m; m >>= 1) v += __shfl_xor(v, m);
  float rstd = rsqrtf(v * (1.0f / DFEAT) + 1e-5f);
  float lg = 0.0f;
  for (int i = lane; i < DFEAT; i += 32)
    lg += (g[i] * (h[i] - mu) * rstd + bta[i]) * Wout[i];
  for (int m = 16; m; m >>= 1) lg += __shfl_xor(lg, m);
  if (lane == 0) logits[row] = lg + bout[0];
}

// ---------------- kernel 6: seq softmax + weighted mean of P ----------------
__global__ void finalize_kernel(const float* logits, const float* P,
                                float* out) {
  int c = blockIdx.x, tid = threadIdx.x;   // 320 threads
  __shared__ float lw[1024];
  __shared__ float red[DFEAT];
  __shared__ float stat;
  for (int i = tid; i < 1024; i += DFEAT) {
    long row = (i < 16) ? ((long)c * 16 + i) : (1024 + (long)c * 1008 + (i - 16));
    lw[i] = logits[row];
  }
  __syncthreads();
  for (int seg = 0; seg < 2; ++seg) {
    int s0 = seg ? 16 : 0, s1 = seg ? 1024 : 16;
    float lm = -3.4e38f;
    for (int i = s0 + tid; i < s1; i += DFEAT) lm = fmaxf(lm, lw[i]);
    red[tid] = lm;
    __syncthreads();
    if (tid == 0) {
      float mx = -3.4e38f;
      for (int j = 0; j < DFEAT; ++j) mx = fmaxf(mx, red[j]);
      stat = mx;
    }
    __syncthreads();
    float mx = stat;
    float ls = 0.0f;
    for (int i = s0 + tid; i < s1; i += DFEAT) {
      float e = __expf(lw[i] - mx); lw[i] = e; ls += e;
    }
    red[tid] = ls;
    __syncthreads();
    if (tid == 0) {
      float ss = 0.0f;
      for (int j = 0; j < DFEAT; ++j) ss += red[j];
      stat = 1.0f / ss;
    }
    __syncthreads();
    float inv = stat;
    for (int i = s0 + tid; i < s1; i += DFEAT) lw[i] *= inv;
    __syncthreads();
  }
  float acc = 0.0f;
  for (int i = 0; i < 1024; ++i) {
    long row = (i < 16) ? ((long)c * 16 + i) : (1024 + (long)c * 1008 + (i - 16));
    acc += lw[i] * P[row * DFEAT + tid];
  }
  out[c * DFEAT + tid] = acc * (1.0f / 1024.0f);
}

// ---------------- launch ----------------------------------------------------
extern "C" void kernel_launch(void* const* d_in, const int* in_sizes, int n_in,
                              void* d_out, int out_size, void* d_ws,
                              size_t ws_size, hipStream_t stream) {
  const float* x_mean = (const float*)d_in[0];
  const float* x      = (const float*)d_in[1];
  const float* W_same = (const float*)d_in[2];
  const float* b_same = (const float*)d_in[3];
  const float* W_diff = (const float*)d_in[4];
  const float* b_diff = (const float*)d_in[5];
  const float* Wq  = (const float*)d_in[6];
  const float* bq  = (const float*)d_in[7];
  const float* Wk  = (const float*)d_in[8];
  const float* bk  = (const float*)d_in[9];
  const float* Wv  = (const float*)d_in[10];
  const float* bv  = (const float*)d_in[11];
  const float* Wfc = (const float*)d_in[12];
  const float* bfc = (const float*)d_in[13];
  const float* ln_g = (const float*)d_in[14];
  const float* ln_b = (const float*)d_in[15];
  const float* Wout = (const float*)d_in[16];
  const float* bout = (const float*)d_in[17];

  char* ws = (char*)d_ws;
  const size_t ALGN = 256;
  auto align_up = [&](size_t v) { return (v + ALGN - 1) & ~(ALGN - 1); };
  size_t off = 0;
  bf16_t* pack = (bf16_t*)(ws + off); off = align_up(off + (size_t)6 * DFEAT * DFEAT * 2);
  float*  P    = (float*) (ws + off); off = align_up(off + (size_t)TOTR * DFEAT * 4);
  bf16_t* Qp   = (bf16_t*)(ws + off); off = align_up(off + (size_t)TOTR * DFEAT * 2);
  bf16_t* Kp   = (bf16_t*)(ws + off); off = align_up(off + (size_t)TOTR * DFEAT * 2);
  bf16_t* Vt   = (bf16_t*)(ws + off); off = align_up(off + (size_t)DFEAT * TOTRP * 2);
  bf16_t* O    = (bf16_t*)(ws + off); off = align_up(off + (size_t)TOTR * DFEAT * 2);
  float*  H    = (float*) (ws + off); off = align_up(off + (size_t)TOTR * DFEAT * 4);
  float*  logits = (float*)(ws + off); off = align_up(off + (size_t)TOTR * 4);

  const int MT = TOTR / 16;                 // 4096 M-tiles
  pack_weights_kernel<<<(6 * KT_N * NT_N * 32 + 255) / 256, 256, 0, stream>>>(
      W_same, W_diff, Wq, Wk, Wv, Wfc, pack);
  init_vt_pad_kernel<<<(DFEAT * 32 + 255) / 256, 256, 0, stream>>>(Vt);
  gemm_p_kernel<<<dim3(MT, 5), 32, 0, stream>>>(x_mean, x, pack, b_same,
                                                b_diff, P);
  gemm_qkv_kernel<<<dim3(MT, 5, 3), 32, 0, stream>>>(P, pack, bq, bk, bv, Qp,
                                                     Kp, Vt);
  attn_kernel<<<64 * 64, 32, 0, stream>>>(Qp, Kp, Vt, O);
  gemm_fc_kernel<<<dim3(MT, 5), 32, 0, stream>>>(O, pack, bfc, P, H);
  ln_logits_kernel<<<TOTR / 8, 256, 0, stream>>>(H, ln_g, ln_b, Wout, bout,
                                                 logits);
  finalize_kernel<<<64, DFEAT, 0, stream>>>(logits, P, (float*)d_out);
}
